// CrossVerseAttention_5987184411236
// MI455X (gfx1250) — matrix operand order repaired and checked by reading, not compiled, not run on hardware
//
#include <hip/hip_runtime.h>

#define B_ 2
#define S_ 2048
#define D_ 1024
#define H_ 16
#define DH_ 64
#define M_ (B_ * S_)            // 4096 tokens
#define SCALE_ 0.125f           // 1/sqrt(64)

typedef __attribute__((ext_vector_type(16))) __bf16 v16bf;
typedef __attribute__((ext_vector_type(8)))  float  v8f;

// ---------------- CDNA5 async global->LDS copy (ASYNCcnt-tracked) ----------
#if defined(__gfx1250__) && __has_builtin(__builtin_amdgcn_global_load_async_to_lds_b128)
#define ASYNC_OK 1
typedef int v4i_ __attribute__((vector_size(16)));
typedef __attribute__((address_space(1))) v4i_ g_v4i;   // global int4
typedef __attribute__((address_space(3))) v4i_ l_v4i;   // LDS int4
#define ASYNC_CP16(dst_lds, src_glb)                                          \
  __builtin_amdgcn_global_load_async_to_lds_b128(                             \
      (g_v4i*)(src_glb), (l_v4i*)(dst_lds), 0, 0)
#if __has_builtin(__builtin_amdgcn_s_wait_asynccnt)
#define WAIT_ASYNC(n) __builtin_amdgcn_s_wait_asynccnt(n)
#else
#define WAIT_ASYNC(n) asm volatile("s_wait_asynccnt %0" ::"i"(n) : "memory")
#endif
#else
#define ASYNC_OK 0
#define WAIT_ASYNC(n) ((void)0)
#endif

__device__ __forceinline__ v8f wmma_bf16(v16bf a, v16bf b, v8f c) {
  return __builtin_amdgcn_wmma_f32_16x16x32_bf16(false, a, false, b, (short)0, c,
                                                 false, false);
}

// ---- WMMA fragment loaders (bf16, K=32; contiguous k = 16*half + e in both A and B) ----
// A-matrix 16x32: lane(0-15)=row M, element pair v: k = 16*half + 2*v
__device__ __forceinline__ v16bf frag_a(const __bf16* base, int stride, int row,
                                        int kOff, int half) {
  v16bf f;
  unsigned* fu = (unsigned*)&f;
  const unsigned short* p = (const unsigned short*)base + (size_t)row * stride + kOff;
#pragma unroll
  for (int v = 0; v < 8; ++v) {
    int k = half * 16 + (v << 1);
    fu[v] = *(const unsigned*)(p + k);
  }
  return f;
}

// B-matrix 32x16, contraction index contiguous in memory:
// lane holds column `colrow` (a stored row); k = 16*half + e
__device__ __forceinline__ v16bf frag_b_rowk(const __bf16* base, int stride,
                                             int colrow, int kOff, int half) {
  v16bf f;
  unsigned* fu = (unsigned*)&f;
  const unsigned short* p =
      (const unsigned short*)base + (size_t)colrow * stride + kOff + half * 16;
#pragma unroll
  for (int v = 0; v < 8; ++v) fu[v] = *(const unsigned*)(p + 2 * v);
  return f;
}

// B-matrix 32x16, contraction index strides across rows: e -> base[(16*half+e)*stride+col]
__device__ __forceinline__ v16bf frag_b_colk(const __bf16* base, int stride, int col,
                                             int half) {
  v16bf f;
  __bf16* fe = (__bf16*)&f;
#pragma unroll
  for (int e = 0; e < 16; ++e)
    fe[e] = base[(size_t)(16 * half + e) * stride + col];
  return f;
}

// ------------------------------- fp32 -> bf16 -------------------------------
__global__ __launch_bounds__(256) void cvt_bf16_kernel(const float* __restrict__ src,
                                                       __bf16* __restrict__ dst,
                                                       size_t n4) {
  size_t i = ((size_t)blockIdx.x * 256 + threadIdx.x);
  if (i >= n4) return;
  float4 f = ((const float4*)src)[i];
  __bf16* d = dst + i * 4;
  d[0] = (__bf16)f.x; d[1] = (__bf16)f.y; d[2] = (__bf16)f.z; d[3] = (__bf16)f.w;
}

// ------------------------------- tiled GEMM --------------------------------
// C[m,n] = sum_k A[m,k] * W[n,k] + bias[n]  (A @ W^T + b)
// mode 0: bf16 out; mode 1: scatter QKV into (B,H,S,DH) x3; mode 2: fp32 out + resid
#define GTM 128
#define GTN 128
#define GTK 32
#define GST 40  // LDS row stride (bf16), 16B-aligned rows, staggers banks

__global__ __launch_bounds__(256) void gemm_bf16_kernel(
    const __bf16* __restrict__ A, const __bf16* __restrict__ W,
    const float* __restrict__ bias, int N, int K, int mode,
    __bf16* __restrict__ outB, float* __restrict__ outF,
    const float* __restrict__ resid) {
  __shared__ __bf16 As[2][GTM * GST];
  __shared__ __bf16 Bs[2][GTN * GST];
  const int tid = threadIdx.x;
  const int lane = tid & 31, wave = tid >> 5;
  const int half = lane >> 4, ln = lane & 15;
  const int wm = wave & 3, wn = wave >> 2;  // 4x2 wave grid -> 32x64 per wave
  const int blockM = blockIdx.y * GTM, blockN = blockIdx.x * GTN;
  const int ldRow = tid >> 1, ldCol = (tid & 1) * 16;

  v8f acc[2][4];
#pragma unroll
  for (int i = 0; i < 2; ++i)
#pragma unroll
    for (int j = 0; j < 4; ++j)
      acc[i][j] = (v8f){0.f, 0.f, 0.f, 0.f, 0.f, 0.f, 0.f, 0.f};

  auto issue = [&](int t, int buf) {
    const __bf16* ga = A + (size_t)(blockM + ldRow) * K + t * GTK + ldCol;
    const __bf16* gb = W + (size_t)(blockN + ldRow) * K + t * GTK + ldCol;
    __bf16* la = &As[buf][ldRow * GST + ldCol];
    __bf16* lb = &Bs[buf][ldRow * GST + ldCol];
#if ASYNC_OK
    ASYNC_CP16(la, ga);
    ASYNC_CP16(la + 8, ga + 8);
    ASYNC_CP16(lb, gb);
    ASYNC_CP16(lb + 8, gb + 8);
#else
    uint4 a0 = ((const uint4*)ga)[0], a1 = ((const uint4*)ga)[1];
    uint4 b0 = ((const uint4*)gb)[0], b1 = ((const uint4*)gb)[1];
    *(uint4*)la = a0; *(uint4*)(la + 8) = a1;
    *(uint4*)lb = b0; *(uint4*)(lb + 8) = b1;
#endif
  };

  const int nT = K / GTK;
  issue(0, 0);
  for (int t = 0; t < nT; ++t) {
    const int cur = t & 1;
    if (t + 1 < nT) {
      issue(t + 1, cur ^ 1);   // prefetch next tile into other buffer
      WAIT_ASYNC(4);           // own tile-t copies done (only next tile in flight)
    } else {
      WAIT_ASYNC(0);
    }
    __syncthreads();           // all waves' tile-t data resident in LDS
    v16bf af[2], bf[4];
#pragma unroll
    for (int mt = 0; mt < 2; ++mt)
      af[mt] = frag_a(As[cur], GST, wm * 32 + mt * 16 + ln, 0, half);
#pragma unroll
    for (int nt = 0; nt < 4; ++nt)
      bf[nt] = frag_b_rowk(Bs[cur], GST, wn * 64 + nt * 16 + ln, 0, half);
#pragma unroll
    for (int mt = 0; mt < 2; ++mt)
#pragma unroll
      for (int nt = 0; nt < 4; ++nt)
        acc[mt][nt] = wmma_bf16(af[mt], bf[nt], acc[mt][nt]);
    __syncthreads();           // reads done before this buffer is refilled at t+2
  }

  const size_t qkvStride = (size_t)B_ * H_ * S_ * DH_;
#pragma unroll
  for (int mt = 0; mt < 2; ++mt)
#pragma unroll
    for (int nt = 0; nt < 4; ++nt)
#pragma unroll
      for (int r = 0; r < 8; ++r) {
        int row = blockM + wm * 32 + mt * 16 + half * 8 + r;
        int col = blockN + wn * 64 + nt * 16 + ln;
        float vl = acc[mt][nt][r] + bias[col];
        if (mode == 0) {
          outB[(size_t)row * N + col] = (__bf16)vl;
        } else if (mode == 1) {
          int which = col >> 10;  // 0=q 1=k 2=v
          int d = col & (D_ - 1);
          int h = d >> 6, dh = d & (DH_ - 1);
          int b = row >> 11, s = row & (S_ - 1);
          __bf16* dst = outB + (size_t)which * qkvStride;
          dst[(((size_t)(b * H_ + h)) * S_ + s) * DH_ + dh] = (__bf16)vl;
        } else {
          outF[(size_t)row * N + col] = vl + resid[(size_t)row * N + col];
        }
      }
}

// ------------------------------ flash attention ----------------------------
__device__ __forceinline__ float maskval(int vi, int vj) {
  int diff = vi - vj;
  diff = diff < 0 ? -diff : diff;
  float m;
  if (diff == 0) m = 1.0f;
  else if (diff == 1) m = 0.7f;
  else m = 0.3f / (float)diff;
  if (vi == 0 || vj == 0) m = 1.0f;
  return m;
}

#define KT_ 32  // keys per iteration
__global__ __launch_bounds__(128) void flash_kernel(
    const __bf16* __restrict__ q, const __bf16* __restrict__ k,
    const __bf16* __restrict__ v, const int* __restrict__ vpos,
    __bf16* __restrict__ ctx /* (B,S,D) */) {
  __shared__ __bf16 Ks[2][KT_ * 72];
  __shared__ __bf16 Vs[2][KT_ * 72];
  __shared__ __bf16 Ps[4 * 16 * 40];
  const int tid = threadIdx.x;
  const int lane = tid & 31, wave = tid >> 5;
  const int half = lane >> 4, ln = lane & 15;
  const int bh = blockIdx.y;  // b*H + h
  const int b = bh >> 4, h = bh & (H_ - 1);
  const int qbase = blockIdx.x * 64 + wave * 16;

  const __bf16* qp = q + (size_t)bh * S_ * DH_;
  const __bf16* kp = k + (size_t)bh * S_ * DH_;
  const __bf16* vpv = v + (size_t)bh * S_ * DH_;
  const int* vposb = vpos + b * S_;

  // Q fragments (16 queries x 64 dh), held in VGPRs for the whole pass
  v16bf qf[2];
#pragma unroll
  for (int c = 0; c < 2; ++c) qf[c] = frag_a(qp, DH_, qbase + ln, c * 32, half);

  int viq[8];
#pragma unroll
  for (int r = 0; r < 8; ++r) viq[r] = vposb[qbase + half * 8 + r];

  v8f cacc[4];
#pragma unroll
  for (int t = 0; t < 4; ++t)
    cacc[t] = (v8f){0.f, 0.f, 0.f, 0.f, 0.f, 0.f, 0.f, 0.f};
  float mrow[8], lrow[8];
#pragma unroll
  for (int r = 0; r < 8; ++r) { mrow[r] = -1e30f; lrow[r] = 0.f; }

  __bf16* P = Ps + wave * (16 * 40);
  const int ldr = tid >> 2, ldc = (tid & 3) * 16;

  auto issueKV = [&](int t, int buf) {
    const __bf16* gk = kp + (size_t)(t * KT_ + ldr) * DH_ + ldc;
    const __bf16* gv = vpv + (size_t)(t * KT_ + ldr) * DH_ + ldc;
    __bf16* lk = &Ks[buf][ldr * 72 + ldc];
    __bf16* lv = &Vs[buf][ldr * 72 + ldc];
#if ASYNC_OK
    ASYNC_CP16(lk, gk);
    ASYNC_CP16(lk + 8, gk + 8);
    ASYNC_CP16(lv, gv);
    ASYNC_CP16(lv + 8, gv + 8);
#else
    uint4 k0 = ((const uint4*)gk)[0], k1 = ((const uint4*)gk)[1];
    uint4 v0 = ((const uint4*)gv)[0], v1 = ((const uint4*)gv)[1];
    *(uint4*)lk = k0; *(uint4*)(lk + 8) = k1;
    *(uint4*)lv = v0; *(uint4*)(lv + 8) = v1;
#endif
  };

  const int nT = S_ / KT_;
  issueKV(0, 0);
  for (int t = 0; t < nT; ++t) {
    const int cur = t & 1;
    const int kt = t * KT_;
    if (t + 1 < nT) {
      issueKV(t + 1, cur ^ 1);
      WAIT_ASYNC(4);
    } else {
      WAIT_ASYNC(0);
    }
    __syncthreads();

    // scores: Q(16x64) x K^T(64x32) -> two 16x16 tiles
    v8f s[2];
#pragma unroll
    for (int sub = 0; sub < 2; ++sub) {
      s[sub] = (v8f){0.f, 0.f, 0.f, 0.f, 0.f, 0.f, 0.f, 0.f};
#pragma unroll
      for (int c = 0; c < 2; ++c) {
        v16bf kb = frag_b_rowk(Ks[cur], 72, sub * 16 + ln, c * 32, half);
        s[sub] = wmma_bf16(qf[c], kb, s[sub]);
      }
    }

    // mask + online softmax (each row's 16 cols live within one 16-lane half)
    int vj0 = vposb[kt + ln];
    int vj1 = vposb[kt + 16 + ln];
    float p0[8], p1[8];
#pragma unroll
    for (int r = 0; r < 8; ++r) {
      int vi = viq[r];
      float s0 = s[0][r] * SCALE_ + maskval(vi, vj0);
      float s1 = s[1][r] * SCALE_ + maskval(vi, vj1);
      float tmx = fmaxf(s0, s1);
#pragma unroll
      for (int mm = 1; mm < 16; mm <<= 1) tmx = fmaxf(tmx, __shfl_xor(tmx, mm, 32));
      float mnew = fmaxf(mrow[r], tmx);
      float alpha = __expf(mrow[r] - mnew);
      float e0 = __expf(s0 - mnew), e1 = __expf(s1 - mnew);
      float psum = e0 + e1;
#pragma unroll
      for (int mm = 1; mm < 16; mm <<= 1) psum += __shfl_xor(psum, mm, 32);
      lrow[r] = lrow[r] * alpha + psum;
      mrow[r] = mnew;
#pragma unroll
      for (int t4 = 0; t4 < 4; ++t4) cacc[t4][r] *= alpha;
      p0[r] = e0; p1[r] = e1;
    }

    // stage P (16x32) through LDS into A-fragment layout (wave-private region)
#pragma unroll
    for (int r = 0; r < 8; ++r) {
      P[(half * 8 + r) * 40 + ln]      = (__bf16)p0[r];
      P[(half * 8 + r) * 40 + 16 + ln] = (__bf16)p1[r];
    }
    asm volatile("s_wait_dscnt 0" ::: "memory");

    v16bf pf = frag_a(P, 40, ln, 0, half);
#pragma unroll
    for (int nt = 0; nt < 4; ++nt) {
      v16bf vf = frag_b_colk(Vs[cur], 72, nt * 16 + ln, half);
      cacc[nt] = wmma_bf16(pf, vf, cacc[nt]);
    }
    __syncthreads();
  }

  // normalize + write ctx in (B,S,D) layout
#pragma unroll
  for (int nt = 0; nt < 4; ++nt)
#pragma unroll
    for (int r = 0; r < 8; ++r) {
      int srow = qbase + half * 8 + r;
      int dh = nt * 16 + ln;
      float o = cacc[nt][r] / lrow[r];
      ctx[((size_t)(b * S_ + srow)) * D_ + h * DH_ + dh] = (__bf16)o;
    }
}

// -------------------------------- layernorm --------------------------------
__global__ __launch_bounds__(256) void ln_kernel(const float* __restrict__ xin,
                                                 const float* __restrict__ gamma,
                                                 const float* __restrict__ beta,
                                                 float* __restrict__ out) {
  __shared__ float red[2];
  __shared__ float rs[8], rq[8];
  const int row = blockIdx.x;
  const float* xr = xin + (size_t)row * D_;
  float vals[4], sum = 0.f, sq = 0.f;
#pragma unroll
  for (int i = 0; i < 4; ++i) {
    float t = xr[threadIdx.x + i * 256];
    vals[i] = t; sum += t; sq += t * t;
  }
#pragma unroll
  for (int mm = 1; mm < 32; mm <<= 1) {
    sum += __shfl_xor(sum, mm, 32);
    sq  += __shfl_xor(sq, mm, 32);
  }
  const int wave = threadIdx.x >> 5, lane = threadIdx.x & 31;
  if (lane == 0) { rs[wave] = sum; rq[wave] = sq; }
  __syncthreads();
  if (wave == 0) {
    float s = lane < 8 ? rs[lane] : 0.f;
    float q2 = lane < 8 ? rq[lane] : 0.f;
#pragma unroll
    for (int mm = 1; mm < 8; mm <<= 1) {
      s += __shfl_xor(s, mm, 32);
      q2 += __shfl_xor(q2, mm, 32);
    }
    if (lane == 0) { red[0] = s; red[1] = q2; }
  }
  __syncthreads();
  float mu = red[0] * (1.0f / D_);
  float var = red[1] * (1.0f / D_) - mu * mu;
  float inv = rsqrtf(var + 1e-5f);
#pragma unroll
  for (int i = 0; i < 4; ++i) {
    int d = threadIdx.x + i * 256;
    out[(size_t)row * D_ + d] = (vals[i] - mu) * inv * gamma[d] + beta[d];
  }
}

// --------------------------------- launch ----------------------------------
extern "C" void kernel_launch(void* const* d_in, const int* in_sizes, int n_in,
                              void* d_out, int out_size, void* d_ws, size_t ws_size,
                              hipStream_t stream) {
  const float* x    = (const float*)d_in[0];
  const int*   vpos = (const int*)d_in[1];
  const float* W_q  = (const float*)d_in[2];
  const float* b_q  = (const float*)d_in[3];
  const float* W_k  = (const float*)d_in[4];
  const float* b_k  = (const float*)d_in[5];
  const float* W_v  = (const float*)d_in[6];
  const float* b_v  = (const float*)d_in[7];
  const float* W_o  = (const float*)d_in[8];
  const float* b_o  = (const float*)d_in[9];
  const float* W_p  = (const float*)d_in[10];
  const float* b_p  = (const float*)d_in[11];
  const float* gamma = (const float*)d_in[12];
  const float* beta  = (const float*)d_in[13];
  float* out = (float*)d_out;

  char* w = (char*)d_ws;
  auto carve = [&](size_t bytes) {
    void* p = (void*)w;
    w += (bytes + 255) & ~(size_t)255;
    return p;
  };
  const size_t MD = (size_t)M_ * D_;
  const size_t DD = (size_t)D_ * D_;
  __bf16* xb    = (__bf16*)carve(MD * 2);
  __bf16* Wqkv  = (__bf16*)carve(3 * DD * 2);
  __bf16* Wob   = (__bf16*)carve(DD * 2);
  __bf16* Wpb   = (__bf16*)carve(DD * 2);
  __bf16* qkv   = (__bf16*)carve(3 * MD * 2);  // q|k|v in (B,H,S,DH)
  __bf16* ctxD  = (__bf16*)carve(MD * 2);
  __bf16* attno = (__bf16*)carve(MD * 2);
  float*  bqkv  = (float*)carve(3 * D_ * 4);
  float*  opre  = (float*)qkv;                 // alias: qkv dead after flash

  // 1) convert activations + weights to bf16
  cvt_bf16_kernel<<<(unsigned)(MD / 1024), 256, 0, stream>>>(x, xb, MD / 4);
  cvt_bf16_kernel<<<(unsigned)(DD / 1024), 256, 0, stream>>>(W_q, Wqkv, DD / 4);
  cvt_bf16_kernel<<<(unsigned)(DD / 1024), 256, 0, stream>>>(W_k, Wqkv + DD, DD / 4);
  cvt_bf16_kernel<<<(unsigned)(DD / 1024), 256, 0, stream>>>(W_v, Wqkv + 2 * DD, DD / 4);
  cvt_bf16_kernel<<<(unsigned)(DD / 1024), 256, 0, stream>>>(W_o, Wob, DD / 4);
  cvt_bf16_kernel<<<(unsigned)(DD / 1024), 256, 0, stream>>>(W_p, Wpb, DD / 4);
  (void)hipMemcpyAsync(bqkv,          b_q, D_ * 4, hipMemcpyDeviceToDevice, stream);
  (void)hipMemcpyAsync(bqkv + D_,     b_k, D_ * 4, hipMemcpyDeviceToDevice, stream);
  (void)hipMemcpyAsync(bqkv + 2 * D_, b_v, D_ * 4, hipMemcpyDeviceToDevice, stream);

  // 2) fused QKV projection: (4096x1024) @ (3072x1024)^T, scatter to (B,H,S,DH)
  gemm_bf16_kernel<<<dim3(3 * D_ / GTN, M_ / GTM), 256, 0, stream>>>(
      xb, Wqkv, bqkv, 3 * D_, D_, /*mode=*/1, qkv, nullptr, nullptr);

  // 3) flash attention with cross-verse mask -> ctx in (B,S,D)
  flash_kernel<<<dim3(S_ / 64, B_ * H_), 128, 0, stream>>>(
      qkv, qkv + MD, qkv + 2 * MD, vpos, ctxD);

  // 4) attn_out = ctx @ W_o^T + b_o  (bf16)
  gemm_bf16_kernel<<<dim3(D_ / GTN, M_ / GTM), 256, 0, stream>>>(
      ctxD, Wob, b_o, D_, D_, /*mode=*/0, attno, nullptr, nullptr);

  // 5) out_pre = x + attn_out @ W_p^T + b_p  (fp32)
  gemm_bf16_kernel<<<dim3(D_ / GTN, M_ / GTM), 256, 0, stream>>>(
      attno, Wpb, b_p, D_, D_, /*mode=*/2, nullptr, opre, x);

  // 6) layernorm -> d_out
  ln_kernel<<<M_, 256, 0, stream>>>(opre, gamma, beta, out);

  (void)in_sizes; (void)n_in; (void)out_size; (void)ws_size;
}
